// ApproachPointPredictor_1941325218390
// MI455X (gfx1250) — matrix operand order repaired and checked by reading, not compiled, not run on hardware
//
#include <hip/hip_runtime.h>
#include <math.h>
#include <stdint.h>

// ---------------------------------------------------------------------------
// CDNA5 (gfx1250) implementation of the PointNet++ FP decoder.
// All Linear layers run on the matrix engine via v_wmma_f32_16x16x32_bf16
// (bf16 operands, f32 accumulate). Weight tiles are staged through LDS and
// shared by the 4 waves of each workgroup (320KB LDS per WGP on MI455X).
// kNN / gather / concat are VALU kernels. Intermediates are bf16 to halve
// HBM traffic (23.3 TB/s roof; workload ~25 GFLOP + ~0.2 GB moved).
// ---------------------------------------------------------------------------

typedef __attribute__((ext_vector_type(16))) __bf16 v16bf;
typedef __attribute__((ext_vector_type(8)))  __bf16 v8bf;
typedef __attribute__((ext_vector_type(4)))  __bf16 v4bf;
typedef __attribute__((ext_vector_type(8)))  float  v8f;

static __device__ __forceinline__ __bf16 f2bf(float f) {
  unsigned u = __builtin_bit_cast(unsigned, f);
  unsigned r = u + 0x7FFFu + ((u >> 16) & 1u);       // round-to-nearest-even
  unsigned short h = (unsigned short)(r >> 16);
  return __builtin_bit_cast(__bf16, h);
}
static __device__ __forceinline__ float bf2f(__bf16 b) {
  unsigned short h = __builtin_bit_cast(unsigned short, b);
  unsigned u = ((unsigned)h) << 16;
  return __builtin_bit_cast(float, u);
}

// ---- weight convert f32[N,K] -> bf16[N,Kpad] (zero pad tail) ---------------
__global__ void cvt_w_kernel(const float* __restrict__ W, __bf16* __restrict__ Wb,
                             int N, int K, int Kpad) {
  int i = blockIdx.x * blockDim.x + threadIdx.x;
  if (i >= N * Kpad) return;
  int n = i / Kpad, k = i % Kpad;
  Wb[i] = (k < K) ? f2bf(W[n * K + k]) : f2bf(0.0f);
}

// ---- fp3 input: broadcast global feature x3[b,G] ++ x2[b,n,C2] -> bf16 -----
__global__ void fp3_concat_kernel(const float* __restrict__ x3, const float* __restrict__ x2,
                                  __bf16* __restrict__ A, int B, int N2, int G, int C2) {
  const int K = G + C2;
  long long i = (long long)blockIdx.x * blockDim.x + threadIdx.x;
  long long total = (long long)B * N2 * K;
  if (i >= total) return;
  int c = (int)(i % K);
  long long r = i / K;
  int b = (int)(r / N2), n = (int)(r % N2);
  float v = (c < G) ? x3[(long long)b * G + c]
                    : x2[((long long)b * N2 + n) * C2 + (c - G)];
  A[i] = f2bf(v);
}

// ---- k=3 nearest neighbors + normalized inverse-d2 weights -----------------
__global__ void knn3_kernel(const float* __restrict__ posSrc, const float* __restrict__ posTgt,
                            int* __restrict__ idxOut, float* __restrict__ wOut,
                            int B, int Ns, int Nt) {
  int t = blockIdx.x * blockDim.x + threadIdx.x;
  if (t >= B * Nt) return;
  int b = t / Nt;
  const float* ps = posSrc + (long long)b * Ns * 3;
  float tx = posTgt[3 * t], ty = posTgt[3 * t + 1], tz = posTgt[3 * t + 2];
  float d0 = 3.0e38f, d1 = 3.0e38f, d2 = 3.0e38f;
  int i0 = 0, i1 = 0, i2 = 0;
  for (int j = 0; j < Ns; ++j) {
    float dx = tx - ps[3 * j], dy = ty - ps[3 * j + 1], dz = tz - ps[3 * j + 2];
    float d = dx * dx + dy * dy + dz * dz;
    if (d < d0)      { d2 = d1; i2 = i1; d1 = d0; i1 = i0; d0 = d; i0 = j; }
    else if (d < d1) { d2 = d1; i2 = i1; d1 = d;  i1 = j; }
    else if (d < d2) { d2 = d;  i2 = j; }
  }
  float w0 = 1.0f / (d0 + 1e-16f), w1 = 1.0f / (d1 + 1e-16f), w2 = 1.0f / (d2 + 1e-16f);
  float s = w0 + w1 + w2;
  idxOut[3 * t] = i0; idxOut[3 * t + 1] = i1; idxOut[3 * t + 2] = i2;
  wOut[3 * t] = w0 / s; wOut[3 * t + 1] = w1 / s; wOut[3 * t + 2] = w2 / s;
}

// ---- weighted 3-NN gather of bf16 features ++ f32 skip feats -> bf16 A -----
__global__ void interp_concat_kernel(const __bf16* __restrict__ F, const int* __restrict__ idx,
                                     const float* __restrict__ wgt, const float* __restrict__ skip,
                                     __bf16* __restrict__ A,
                                     int B, int Ns, int Nt, int C1, int C2, int Kpad) {
  long long i = (long long)blockIdx.x * blockDim.x + threadIdx.x;
  long long total = (long long)B * Nt * Kpad;
  if (i >= total) return;
  int c = (int)(i % Kpad);
  long long r = i / Kpad;
  int b = (int)(r / Nt);
  float v;
  if (c < C1) {
    const int*   id = idx + 3 * r;
    const float* ww = wgt + 3 * r;
    long long base = (long long)b * Ns;
    v = ww[0] * bf2f(F[(base + id[0]) * C1 + c])
      + ww[1] * bf2f(F[(base + id[1]) * C1 + c])
      + ww[2] * bf2f(F[(base + id[2]) * C1 + c]);
  } else if (c < C1 + C2) {
    v = skip[r * C2 + (c - C1)];
  } else {
    v = 0.0f;
  }
  A[i] = f2bf(v);
}

// ---- WMMA GEMM: C[M,N] = epi(A[M,K] @ W[N,K]^T) ----------------------------
// Block = 128 threads (4 waves) computes a 64(M) x 16(N) strip; the 32x16
// weight tile for the shared N-tile is staged in LDS per K-step (coalesced
// global loads) and read back per-lane via ds_load_b128 in the WMMA B layout.
// M must be a multiple of 64 and N of 16 (true for every layer here), so all
// waves participate in every barrier and EXEC is all-ones at each WMMA.
// Epilogue: (acc + bias[n]) * (gamma[n]*bn_inv) + beta[n], optional ReLU.
__global__ void __launch_bounds__(128)
wmma_gemm_bf16_kernel(const __bf16* __restrict__ A, const __bf16* __restrict__ W,
                      const float* __restrict__ bias, const float* __restrict__ gamma,
                      const float* __restrict__ beta, float bn_inv,
                      __bf16* __restrict__ C, int M, int N, int K, int relu) {
  __shared__ __bf16 sW[16][32];                           // 1 KB weight tile

  const int t    = threadIdx.x;
  const int lane = t & 31;
  const int wv   = t >> 5;                                // wave id: M subtile
  const int tm   = (blockIdx.x << 2) + wv;
  const int mrow = (tm << 4) + (lane & 15);
  const int ncol = (blockIdx.y << 4) + (lane & 15);
  const int ksel = lane >> 4;                             // half-wave selector

  const __bf16* Arow = A + (long long)mrow * K;
  // staging assignment: thread t copies 4 bf16 of row (n = t/8), k-chunk t%8
  const int sn = t >> 3;
  const int sk = (t & 7) << 2;
  const __bf16* Wsrc = W + (long long)(((blockIdx.y << 4) + sn)) * K + sk;

  union { v16bf v; v8bf h[2]; } aU, bU;
  v8f acc = {};
  for (int k0 = 0; k0 < K; k0 += 32) {
    // stage the 32x16 weight tile into LDS (8B per thread, coalesced)
    *(v4bf*)&sW[sn][sk] = *(const v4bf*)(Wsrc + k0);
    // A 16x32 fragment direct from global (per-wave rows; no cross-wave reuse)
    aU.h[0] = *(const v8bf*)(Arow + k0 + ksel * 8);
    aU.h[1] = *(const v8bf*)(Arow + k0 + 16 + ksel * 8);
    if (k0 + 32 < K) __builtin_prefetch(Arow + k0 + 32, 0, 0); // global_prefetch_b8
    __syncthreads();
    // B 32x16 fragment from LDS: lanes<16 hold K 0..7,16..23 of col n, >=16 K 8..15,24..31
    bU.h[0] = *(const v8bf*)&sW[lane & 15][ksel * 8];
    bU.h[1] = *(const v8bf*)&sW[lane & 15][16 + ksel * 8];
    __syncthreads();                                      // reads done before restage
    acc = __builtin_amdgcn_wmma_f32_16x16x32_bf16(false, aU.v, false, bU.v,
                                                  (short)0, acc, false, false);
  }

  const float bi = bias  ? bias[ncol]           : 0.0f;
  const float sc = gamma ? gamma[ncol] * bn_inv : 1.0f;
  const float sh = beta  ? beta[ncol]           : 0.0f;
  const int rbase = (tm << 4) + (ksel << 3);              // C/D rows r / r+8
  #pragma unroll
  for (int r = 0; r < 8; ++r) {
    float v = (acc[r] + bi) * sc + sh;
    if (relu) v = fmaxf(v, 0.0f);
    C[(long long)(rbase + r) * N + ncol] = f2bf(v);
  }
}

// ---- head tail: out = sigmoid(h[64] . w1 + b1) -----------------------------
__global__ void head_final_kernel(const __bf16* __restrict__ H, const float* __restrict__ w1,
                                  const float* __restrict__ b1, float* __restrict__ out, int M) {
  int p = blockIdx.x * blockDim.x + threadIdx.x;
  if (p >= M) return;
  const __bf16* h = H + (long long)p * 64;
  float acc = b1[0];
  #pragma unroll
  for (int c = 0; c < 64; ++c) acc += w1[c] * bf2f(h[c]);
  out[p] = 1.0f / (1.0f + expf(-acc));
}

// ---------------------------------------------------------------------------
extern "C" void kernel_launch(void* const* d_in, const int* in_sizes, int n_in,
                              void* d_out, int out_size, void* d_ws, size_t ws_size,
                              hipStream_t stream) {
  (void)in_sizes; (void)n_in; (void)ws_size; (void)out_size;
  const int B = 32, N0 = 4096, N1 = 1024, N2 = 256, G = 1024;
  const int M3 = B * N2, M2 = B * N1, M1 = B * N0;
  const float BN_INV = 1.0f / sqrtf(1.0f + 1e-5f);

  const float* x3   = (const float*)d_in[0];
  const float* x2   = (const float*)d_in[2];
  const float* pos2 = (const float*)d_in[3];
  const float* x1   = (const float*)d_in[4];
  const float* pos1 = (const float*)d_in[5];
  const float* x0   = (const float*)d_in[6];
  const float* pos0 = (const float*)d_in[7];
  const float* fp3_W0 = (const float*)d_in[8],  *fp3_b0 = (const float*)d_in[9];
  const float* fp3_g0 = (const float*)d_in[10], *fp3_be0 = (const float*)d_in[11];
  const float* fp3_W1 = (const float*)d_in[12], *fp3_b1 = (const float*)d_in[13];
  const float* fp2_W0 = (const float*)d_in[14], *fp2_b0 = (const float*)d_in[15];
  const float* fp2_g0 = (const float*)d_in[16], *fp2_be0 = (const float*)d_in[17];
  const float* fp2_W1 = (const float*)d_in[18], *fp2_b1 = (const float*)d_in[19];
  const float* fp1_W0 = (const float*)d_in[20], *fp1_b0 = (const float*)d_in[21];
  const float* fp1_g0 = (const float*)d_in[22], *fp1_be0 = (const float*)d_in[23];
  const float* fp1_W1 = (const float*)d_in[24], *fp1_b1 = (const float*)d_in[25];
  const float* fp1_g1 = (const float*)d_in[26], *fp1_be1 = (const float*)d_in[27];
  const float* fp1_W2 = (const float*)d_in[28], *fp1_b2 = (const float*)d_in[29];
  const float* head_W0 = (const float*)d_in[30], *head_b0 = (const float*)d_in[31];
  const float* head_W1 = (const float*)d_in[32], *head_b1 = (const float*)d_in[33];

  // ---- workspace carve (deterministic) -----------------------------------
  uint8_t* ws = (uint8_t*)d_ws;
  size_t off = 0;
  auto alloc = [&](size_t bytes) -> void* {
    void* p = ws + off;
    off += (bytes + 255) & ~(size_t)255;
    return p;
  };
  __bf16* Wb3_0 = (__bf16*)alloc((size_t)256 * 1280 * 2);
  __bf16* Wb3_1 = (__bf16*)alloc((size_t)256 * 256 * 2);
  __bf16* Wb2_0 = (__bf16*)alloc((size_t)256 * 384 * 2);
  __bf16* Wb2_1 = (__bf16*)alloc((size_t)128 * 256 * 2);
  __bf16* Wb1_0 = (__bf16*)alloc((size_t)128 * 160 * 2);   // 131 -> 160 padded
  __bf16* Wb1_1 = (__bf16*)alloc((size_t)128 * 128 * 2);
  __bf16* Wb1_2 = (__bf16*)alloc((size_t)128 * 128 * 2);
  __bf16* WbH0  = (__bf16*)alloc((size_t)64 * 128 * 2);
  int*    idx2  = (int*)   alloc((size_t)M2 * 3 * 4);
  float*  w2    = (float*) alloc((size_t)M2 * 3 * 4);
  int*    idx1  = (int*)   alloc((size_t)M1 * 3 * 4);
  float*  w1    = (float*) alloc((size_t)M1 * 3 * 4);
  __bf16* Abuf  = (__bf16*)alloc((size_t)M1 * 160 * 2);    // >= 8192*1280, 32768*384
  __bf16* H1    = (__bf16*)alloc((size_t)M1 * 128 * 2);    // >= 32768*256
  __bf16* H2    = (__bf16*)alloc((size_t)M1 * 128 * 2);

  auto ew = [&](long long total) { return dim3((unsigned)((total + 255) / 256)); };
  auto gemm = [&](const __bf16* Aa, const __bf16* Ww, const float* bi, const float* g,
                  const float* be, __bf16* Cc, int Mm, int Nn, int Kk, int relu) {
    dim3 grid((unsigned)(Mm / 64), (unsigned)(Nn / 16));
    wmma_gemm_bf16_kernel<<<grid, dim3(128), 0, stream>>>(
        Aa, Ww, bi, g, be, BN_INV, Cc, Mm, Nn, Kk, relu);
  };

  // ---- weight conversions -------------------------------------------------
  cvt_w_kernel<<<ew(256LL * 1280), 256, 0, stream>>>(fp3_W0, Wb3_0, 256, 1280, 1280);
  cvt_w_kernel<<<ew(256LL * 256),  256, 0, stream>>>(fp3_W1, Wb3_1, 256, 256, 256);
  cvt_w_kernel<<<ew(256LL * 384),  256, 0, stream>>>(fp2_W0, Wb2_0, 256, 384, 384);
  cvt_w_kernel<<<ew(128LL * 256),  256, 0, stream>>>(fp2_W1, Wb2_1, 128, 256, 256);
  cvt_w_kernel<<<ew(128LL * 160),  256, 0, stream>>>(fp1_W0, Wb1_0, 128, 131, 160);
  cvt_w_kernel<<<ew(128LL * 128),  256, 0, stream>>>(fp1_W1, Wb1_1, 128, 128, 128);
  cvt_w_kernel<<<ew(128LL * 128),  256, 0, stream>>>(fp1_W2, Wb1_2, 128, 128, 128);
  cvt_w_kernel<<<ew(64LL * 128),   256, 0, stream>>>(head_W0, WbH0, 64, 128, 128);

  // ---- fp3: broadcast-concat -> 1280->256 (BN+ReLU) -> 256->256 -----------
  fp3_concat_kernel<<<ew((long long)M3 * 1280), 256, 0, stream>>>(x3, x2, Abuf, B, N2, G, 256);
  gemm(Abuf, Wb3_0, fp3_b0, fp3_g0, fp3_be0, H1, M3, 256, 1280, 1);
  gemm(H1, Wb3_1, fp3_b1, nullptr, nullptr, H2, M3, 256, 256, 0);   // H2 = fp3 feats

  // ---- fp2: knn3(256 -> 1024), gather+concat, 384->256 (BN+ReLU), 256->128
  knn3_kernel<<<ew(M2), 256, 0, stream>>>(pos2, pos1, idx2, w2, B, N2, N1);
  interp_concat_kernel<<<ew((long long)M2 * 384), 256, 0, stream>>>(
      H2, idx2, w2, x1, Abuf, B, N2, N1, 256, 128, 384);
  gemm(Abuf, Wb2_0, fp2_b0, fp2_g0, fp2_be0, H1, M2, 256, 384, 1);
  gemm(H1, Wb2_1, fp2_b1, nullptr, nullptr, H2, M2, 128, 256, 0);   // H2 = fp2 feats

  // ---- fp1: knn3(1024 -> 4096), gather+concat(pad 131->160), 3 GEMMs ------
  knn3_kernel<<<ew(M1), 256, 0, stream>>>(pos1, pos0, idx1, w1, B, N1, N0);
  interp_concat_kernel<<<ew((long long)M1 * 160), 256, 0, stream>>>(
      H2, idx1, w1, x0, Abuf, B, N1, N0, 128, 3, 160);
  gemm(Abuf, Wb1_0, fp1_b0, fp1_g0, fp1_be0, H1, M1, 128, 160, 1);
  gemm(H1, Wb1_1, fp1_b1, fp1_g1, fp1_be1, H2, M1, 128, 128, 1);
  gemm(H2, Wb1_2, fp1_b2, nullptr, nullptr, H1, M1, 128, 128, 0);

  // ---- head: 128->64 (ReLU) on WMMA, then 64->1 + sigmoid -----------------
  gemm(H1, WbH0, head_b0, nullptr, nullptr, H2, M1, 64, 128, 1);
  head_final_kernel<<<ew(M1), 256, 0, stream>>>(H2, head_W1, head_b1, (float*)d_out, M1);
}
